// SegmentedAttention_16338055594765
// MI455X (gfx1250) — matrix operand order repaired and checked by reading, not compiled, not run on hardware
//
#include <hip/hip_runtime.h>

// ---------------- CDNA5 / gfx1250 types ----------------
typedef __attribute__((ext_vector_type(16))) __bf16 bf16x16;
typedef __attribute__((ext_vector_type(8)))  __bf16 bf16x8;
typedef __attribute__((ext_vector_type(8)))  float  f32x8;

#define DIM    1024
#define HEADS  16
#define DHEAD  64
#define SEG    512
#define PMEM   16
#define KPAD   544              // PMEM + SEG + 16 zero-pad rows -> multiple of 32
#define NTOK   8192             // 2 * 4096 tokens
#define NSEG   16               // (2*4096)/512 folded segment-batches
#define NQKV   3072             // 3 * HEADS * DHEAD

__device__ __forceinline__ __bf16 f2bf(float f) {
  unsigned u = __builtin_bit_cast(unsigned, f);
  u += 0x7FFFu + ((u >> 16) & 1u);              // round-to-nearest-even
  unsigned short h = (unsigned short)(u >> 16);
  return __builtin_bit_cast(__bf16, h);
}

// A-fragment (16x32 bf16): lane m = lane&15, half = lane>>4.
// elements 0..7 = row[m][half*16 + 0..7], 8..15 = row[m][half*16 + 8..15]
__device__ __forceinline__ bf16x16 ldafrag(const __bf16* p0, const __bf16* p1) {
  bf16x8 lo = *(const bf16x8*)p0;
  bf16x8 hi = *(const bf16x8*)p1;
  bf16x16 a;
#pragma unroll
  for (int i = 0; i < 8; ++i) { a[i] = lo[i]; a[8 + i] = hi[i]; }
  return a;
}

// B-fragment (32x16 bf16, B(k,n) = row-major source row n): lane n = lane&15,
// half = lane>>4; b[i] = srcrow_n[k0 + half*16 + i]  (16 contiguous bf16)
__device__ __forceinline__ bf16x16 ldbfrag(const __bf16* p) {
  bf16x8 lo = *(const bf16x8*)p;
  bf16x8 hi = *(const bf16x8*)(p + 8);
  bf16x16 b;
#pragma unroll
  for (int i = 0; i < 8; ++i) { b[i] = lo[i]; b[8 + i] = hi[i]; }
  return b;
}

__device__ __forceinline__ f32x8 wmma_bf16(bf16x16 a, bf16x16 b, f32x8 c) {
  return __builtin_amdgcn_wmma_f32_16x16x32_bf16(false, a, false, b, (short)0, c,
                                                 false, false);
}

// ---------------- elementwise: f32 -> bf16 ----------------
__global__ void k_f32_to_bf16(const float* __restrict__ src,
                              __bf16* __restrict__ dst, int n) {
  for (int i = blockIdx.x * blockDim.x + threadIdx.x; i < n;
       i += gridDim.x * blockDim.x)
    dst[i] = f2bf(src[i]);
}

// ---------------- RMSNorm (one block per token row) ----------------
__global__ __launch_bounds__(256) void k_rmsnorm(const float* __restrict__ seq,
                                                 const float* __restrict__ g,
                                                 __bf16* __restrict__ xb) {
  int row = blockIdx.x;
  int tid = threadIdx.x;
  const float* x = seq + (size_t)row * DIM;
  float ss = 0.f;
  for (int i = tid; i < DIM; i += 256) { float v = x[i]; ss += v * v; }
#pragma unroll
  for (int off = 16; off >= 1; off >>= 1) ss += __shfl_xor(ss, off, 32);
  __shared__ float red[8];
  if ((tid & 31) == 0) red[tid >> 5] = ss;
  __syncthreads();
  float tot = 0.f;
#pragma unroll
  for (int i = 0; i < 8; ++i) tot += red[i];
  float scl = rsqrtf(tot * (1.0f / DIM) + 1.1920929e-07f);
  __bf16* o = xb + (size_t)row * DIM;
  for (int i = tid; i < DIM; i += 256) o[i] = f2bf(x[i] * scl * g[i]);
}

// ------- fused QKV GEMM + RoPE + attention-layout staging (WMMA) -----------
// qkv[t, c] = Xb[t,:] @ Wq[c,:];  c<1024 -> Q (RoPE), <2048 -> K (RoPE),
// else -> V.  Writes bf16 directly to Qb / Kb (PMEM offset) / Vt (transposed),
// eliminating the f32 qkv intermediate (saves ~250 MB of HBM traffic).
__global__ __launch_bounds__(256) void k_gemm_qkv_rope(
    const __bf16* __restrict__ A, const __bf16* __restrict__ W,
    __bf16* __restrict__ Qb, __bf16* __restrict__ Kb, __bf16* __restrict__ Vt) {
  const int K = DIM;
  int lane = threadIdx.x & 31;
  int wv = threadIdx.x >> 5;
  int wm = wv & 3, wn = wv >> 2;
  int row0 = blockIdx.y * 128 + wm * 32;   // token rows
  int col0 = blockIdx.x * 128 + wn * 64;   // qkv feature cols
  int ml = lane & 15, h = lane >> 4;

  f32x8 acc[2][4];
#pragma unroll
  for (int i = 0; i < 2; ++i)
#pragma unroll
    for (int j = 0; j < 4; ++j)
#pragma unroll
      for (int e = 0; e < 8; ++e) acc[i][j][e] = 0.f;

  for (int k0 = 0; k0 < K; k0 += 32) {
    if (k0 + 128 < K) {
      __builtin_prefetch(A + (size_t)(row0 + ml) * K + k0 + 128, 0, 1);
      __builtin_prefetch(W + (size_t)(col0 + ml) * K + k0 + 128, 0, 1);
    }
    bf16x16 a[2], b[4];
#pragma unroll
    for (int mt = 0; mt < 2; ++mt) {
      const __bf16* r = A + (size_t)(row0 + mt * 16 + ml) * K + k0;
      a[mt] = ldafrag(r + h * 16, r + h * 16 + 8);
    }
#pragma unroll
    for (int nt = 0; nt < 4; ++nt) {
      const __bf16* r = W + (size_t)(col0 + nt * 16 + ml) * K + k0 + h * 16;
      b[nt] = ldbfrag(r);
    }
#pragma unroll
    for (int mt = 0; mt < 2; ++mt)
#pragma unroll
      for (int nt = 0; nt < 4; ++nt)
        acc[mt][nt] = wmma_bf16(a[mt], b[nt], acc[mt][nt]);
  }

  // Epilogue: C/D layout row M = v + 8*half (token), col N = lane&15.
  // RoPE partner (interleaved pairs) = adjacent feature column = lane^1.
  int sec = col0 >> 10;                      // uniform per block: 0=q,1=k,2=v
#pragma unroll
  for (int nt = 0; nt < 4; ++nt) {
    int gcol = col0 + nt * 16 + ml;
    int d = gcol & 63;
    int hd = (gcol >> 6) & 15;
    float invf = __powf(10000.f, -(float)(2 * (d >> 1)) * (1.f / 64.f));
    float sgn = (d & 1) ? 1.f : -1.f;        // rotate_half interleaved
#pragma unroll
    for (int mt = 0; mt < 2; ++mt) {
#pragma unroll
      for (int v = 0; v < 8; ++v) {
        int t = row0 + mt * 16 + v + 8 * h;  // token index
        float val = acc[mt][nt][v];
        float partner = __shfl_xor(val, 1, 32);
        int sg = t >> 9, qi = t & 511;
        size_t sh = (size_t)(sg * HEADS + hd);
        if (sec == 2) {
          Vt[(sh * DHEAD + d) * KPAD + PMEM + qi] = f2bf(val);
        } else {
          float pos = (float)(t & 4095);     // full-sequence RoPE position
          float sn, cs;
          __sincosf(pos * invf, &sn, &cs);
          float r = val * cs + sgn * partner * sn;
          if (sec == 0)
            Qb[(sh * SEG + qi) * DHEAD + d] = f2bf(r);
          else
            Kb[(sh * KPAD + PMEM + qi) * DHEAD + d] = f2bf(r);
        }
      }
    }
  }
}

// ---------------- generic WMMA GEMM: C[M,N] f32 = A @ W^T ----------------
__global__ __launch_bounds__(256) void k_gemm_bf16(
    const __bf16* __restrict__ A, const __bf16* __restrict__ W,
    float* __restrict__ C, int M, int N, int K) {
  int lane = threadIdx.x & 31;
  int wv = threadIdx.x >> 5;
  int wm = wv & 3, wn = wv >> 2;
  int row0 = blockIdx.y * 128 + wm * 32;
  int col0 = blockIdx.x * 128 + wn * 64;
  int ml = lane & 15, h = lane >> 4;

  f32x8 acc[2][4];
#pragma unroll
  for (int i = 0; i < 2; ++i)
#pragma unroll
    for (int j = 0; j < 4; ++j)
#pragma unroll
      for (int e = 0; e < 8; ++e) acc[i][j][e] = 0.f;

  for (int k0 = 0; k0 < K; k0 += 32) {
    if (k0 + 128 < K) {
      __builtin_prefetch(A + (size_t)(row0 + ml) * K + k0 + 128, 0, 1);
      __builtin_prefetch(W + (size_t)(col0 + ml) * K + k0 + 128, 0, 1);
    }
    bf16x16 a[2], b[4];
#pragma unroll
    for (int mt = 0; mt < 2; ++mt) {
      const __bf16* r = A + (size_t)(row0 + mt * 16 + ml) * K + k0;
      a[mt] = ldafrag(r + h * 16, r + h * 16 + 8);
    }
#pragma unroll
    for (int nt = 0; nt < 4; ++nt) {
      const __bf16* r = W + (size_t)(col0 + nt * 16 + ml) * K + k0 + h * 16;
      b[nt] = ldbfrag(r);
    }
#pragma unroll
    for (int mt = 0; mt < 2; ++mt)
#pragma unroll
      for (int nt = 0; nt < 4; ++nt)
        acc[mt][nt] = wmma_bf16(a[mt], b[nt], acc[mt][nt]);
  }
#pragma unroll
  for (int mt = 0; mt < 2; ++mt)
#pragma unroll
    for (int nt = 0; nt < 4; ++nt)
#pragma unroll
      for (int v = 0; v < 8; ++v)
        C[(size_t)(row0 + mt * 16 + v + 8 * h) * N + col0 + nt * 16 + ml] =
            acc[mt][nt][v];
}

// -------- persistent-memory rows + zero padding rows for K / Vt ----------
__global__ __launch_bounds__(256) void k_stage_pm(const float* __restrict__ pm,
                                                  __bf16* __restrict__ Kb,
                                                  __bf16* __restrict__ Vt) {
  int idx = blockIdx.x * 256 + threadIdx.x;      // [0, 16*16*32*64)
  int d = idx & 63;
  int j = (idx >> 6) & 31;
  int hd = (idx >> 11) & 15;
  int sg = idx >> 15;
  size_t sh = (size_t)(sg * HEADS + hd);
  if (j < PMEM) {
    float kv = pm[((size_t)hd * PMEM + j) * DHEAD + d];                 // pm[0]
    float vv = pm[(((size_t)HEADS + hd) * PMEM + j) * DHEAD + d];       // pm[1]
    Kb[(sh * KPAD + j) * DHEAD + d] = f2bf(kv);
    Vt[(sh * DHEAD + d) * KPAD + j] = f2bf(vv);
  } else {
    int jj = 512 + j;                            // pad rows 528..543
    Kb[(sh * KPAD + jj) * DHEAD + d] = f2bf(0.f);
    Vt[(sh * DHEAD + d) * KPAD + jj] = f2bf(0.f);
  }
}

// ---------------- flash attention: 1 wave per (seg, head, 16-query tile) ---
__global__ __launch_bounds__(32) void k_attention(const __bf16* __restrict__ Qb,
                                                  const __bf16* __restrict__ Kb,
                                                  const __bf16* __restrict__ Vt,
                                                  __bf16* __restrict__ Ob) {
  int qt = blockIdx.x, hd = blockIdx.y, sg = blockIdx.z;
  int lane = threadIdx.x;
  int nl = lane & 15, h = lane >> 4;
  size_t sh = (size_t)(sg * HEADS + hd);
  const __bf16* Qh = Qb + sh * SEG * DHEAD;
  const __bf16* Kh = Kb + sh * KPAD * DHEAD;
  const __bf16* Vh = Vt + sh * DHEAD * KPAD;
  int qr0 = qt * 16;

  // Q A-fragments for d=0..31 and d=32..63
  bf16x16 qlo, qhi;
  {
    const __bf16* r = Qh + (size_t)(qr0 + nl) * DHEAD;
    qlo = ldafrag(r + h * 16, r + h * 16 + 8);
    qhi = ldafrag(r + 32 + h * 16, r + 40 + h * 16);
  }

  f32x8 O[4];
  float mst[8], lst[8], alp[8];
#pragma unroll
  for (int nt = 0; nt < 4; ++nt)
#pragma unroll
    for (int e = 0; e < 8; ++e) O[nt][e] = 0.f;
#pragma unroll
  for (int v = 0; v < 8; ++v) { mst[v] = -3.0e38f; lst[v] = 0.f; }

  __shared__ __align__(16) __bf16 Pl[16 * 32];
  const float scale = 0.125f;                    // 64^-0.5

  for (int j0 = 0; j0 < KPAD; j0 += 32) {        // 17 key blocks of 32
    if (j0 + 32 < KPAD) {                        // prefetch next key block
      __builtin_prefetch(Kh + (size_t)(j0 + 32 + nl) * DHEAD, 0, 1);
      __builtin_prefetch(Vh + (size_t)nl * KPAD + j0 + 32, 0, 1);
    }
    f32x8 S[2];
#pragma unroll
    for (int c = 0; c < 2; ++c)
#pragma unroll
      for (int e = 0; e < 8; ++e) S[c][e] = 0.f;

#pragma unroll
    for (int c = 0; c < 2; ++c) {
      const __bf16* kr = Kh + (size_t)(j0 + c * 16 + nl) * DHEAD;
      S[c] = wmma_bf16(qlo, ldbfrag(kr + h * 16), S[c]);
      S[c] = wmma_bf16(qhi, ldbfrag(kr + 32 + h * 16), S[c]);
    }

    // online softmax over the 32 keys; stats per row (8 rows per half-wave)
#pragma unroll
    for (int v = 0; v < 8; ++v) {
      int qi = qr0 + v + 8 * h;
      float s0 = S[0][v] * scale;
      if (j0 + nl > qi + PMEM) s0 = -1e30f;
      float s1 = S[1][v] * scale;
      if (j0 + 16 + nl > qi + PMEM) s1 = -1e30f;
      float t = fmaxf(s0, s1);
#pragma unroll
      for (int off = 1; off < 16; off <<= 1) t = fmaxf(t, __shfl_xor(t, off, 32));
      float mn = fmaxf(mst[v], t);
      float al = __expf(mst[v] - mn);
      float p0 = __expf(s0 - mn), p1 = __expf(s1 - mn);
      float rs = p0 + p1;
#pragma unroll
      for (int off = 1; off < 16; off <<= 1) rs += __shfl_xor(rs, off, 32);
      lst[v] = lst[v] * al + rs;
      mst[v] = mn;
      alp[v] = al;
      Pl[(v + 8 * h) * 32 + nl] = f2bf(p0);
      Pl[(v + 8 * h) * 32 + 16 + nl] = f2bf(p1);
    }
#pragma unroll
    for (int nt = 0; nt < 4; ++nt)
#pragma unroll
      for (int v = 0; v < 8; ++v) O[nt][v] *= alp[v];
    __syncthreads();

    // reload P in A-fragment layout from LDS, then O += P @ V
    bf16x16 pa;
    {
      const __bf16* r = &Pl[0] + nl * 32;
      pa = ldafrag(r + h * 16, r + h * 16 + 8);
    }
#pragma unroll
    for (int nt = 0; nt < 4; ++nt) {
      const __bf16* vr = Vh + (size_t)(nt * 16 + nl) * KPAD + j0 + h * 16;
      O[nt] = wmma_bf16(pa, ldbfrag(vr), O[nt]);
    }
    __syncthreads();
  }

  // normalize and write merged-head bf16 output row = sg*512 + qi, col = hd*64+d
#pragma unroll
  for (int v = 0; v < 8; ++v) {
    float inv = 1.f / lst[v];
    size_t row = (size_t)sg * SEG + qr0 + v + 8 * h;
#pragma unroll
    for (int nt = 0; nt < 4; ++nt)
      Ob[row * (HEADS * DHEAD) + hd * DHEAD + nt * 16 + nl] =
          f2bf(O[nt][v] * inv);
  }
}

// ---------------- launcher ----------------
extern "C" void kernel_launch(void* const* d_in, const int* in_sizes, int n_in,
                              void* d_out, int out_size, void* d_ws,
                              size_t ws_size, hipStream_t stream) {
  (void)in_sizes; (void)n_in; (void)out_size; (void)ws_size;
  const float* seq  = (const float*)d_in[0];
  const float* g    = (const float*)d_in[1];
  const float* wqkv = (const float*)d_in[2];
  const float* wout = (const float*)d_in[3];
  const float* pm   = (const float*)d_in[4];
  float* out = (float*)d_out;

  char* ws = (char*)d_ws;
  size_t off = 0;
  auto carve = [&](size_t bytes) {
    size_t o = off;
    off += (bytes + 255) & ~(size_t)255;
    return o;
  };
  __bf16* Xb  = (__bf16*)(ws + carve((size_t)NTOK * DIM * 2));
  __bf16* Wqb = (__bf16*)(ws + carve((size_t)NQKV * DIM * 2));
  __bf16* Wob = (__bf16*)(ws + carve((size_t)DIM * DIM * 2));
  __bf16* Qb  = (__bf16*)(ws + carve((size_t)NSEG * HEADS * SEG * DHEAD * 2));
  __bf16* Kb  = (__bf16*)(ws + carve((size_t)NSEG * HEADS * KPAD * DHEAD * 2));
  __bf16* Vt  = (__bf16*)(ws + carve((size_t)NSEG * HEADS * DHEAD * KPAD * 2));
  __bf16* Ob  = (__bf16*)(ws + carve((size_t)NTOK * DIM * 2));

  // 1) weights -> bf16
  k_f32_to_bf16<<<2048, 256, 0, stream>>>(wqkv, Wqb, NQKV * DIM);
  k_f32_to_bf16<<<1024, 256, 0, stream>>>(wout, Wob, DIM * DIM);
  // 2) RMSNorm -> bf16 activations
  k_rmsnorm<<<NTOK, 256, 0, stream>>>(seq, g, Xb);
  // 3) persistent-mem rows + zero pads for K / Vt
  k_stage_pm<<<(NSEG * HEADS * 32 * 64) / 256, 256, 0, stream>>>(pm, Kb, Vt);
  // 4) fused QKV projection + RoPE + staging (WMMA)
  k_gemm_qkv_rope<<<dim3(NQKV / 128, NTOK / 128), 256, 0, stream>>>(
      Xb, Wqb, Qb, Kb, Vt);
  // 5) flash attention (WMMA, online softmax)
  k_attention<<<dim3(SEG / 16, HEADS, NSEG), 32, 0, stream>>>(Qb, Kb, Vt, Ob);
  // 6) output projection (WMMA) -> f32 d_out
  k_gemm_bf16<<<dim3(DIM / 128, NTOK / 128), 256, 0, stream>>>(
      Ob, Wob, out, NTOK, DIM, DIM);
}
